// DetectionLoss_85212151152986
// MI455X (gfx1250) — matrix operand order repaired and checked by reading, not compiled, not run on hardware
//
#include <hip/hip_runtime.h>
#include <math.h>

#define BB   16
#define MM   32
#define NN   8400
#define NCC  80
#define RMM  16
#define EPSF 1e-7f
#define TPB  256

typedef __attribute__((ext_vector_type(16))) _Float16 v16h;
typedef __attribute__((ext_vector_type(8)))  float    v8f;

// ---------------------------------------------------------------------------
// Kernel 1: DFL decode via WMMA.
// One wave32 handles 4 anchor points = 16 softmax-16 distributions.
// Stage 1: all 32 lanes cooperatively load the wave's 256 contiguous floats
//          (2x float4 per lane -> global_load_b128, fully coalesced) into LDS.
// Stage 2: lanes 0..15 softmax their 16-float row, store f16 probs to LDS.
// Stage 3: WMMA f32_16x16x32_f16: A rows = distributions (K 16..31 zero),
//          B[k][j] = k for k<16 (projection vector).  D gives d = E[bin].
// ---------------------------------------------------------------------------
__global__ void decode_wmma_kernel(const float* __restrict__ regs,
                                   const float* __restrict__ anchors,
                                   const float* __restrict__ strides,
                                   float* __restrict__ boxes)
{
    __shared__ float    sraw[8][256];        // 1 KB per wave, 16B aligned
    __shared__ _Float16 sprob[8][16][16];
    const int lane    = threadIdx.x & 31;
    const int w       = threadIdx.x >> 5;
    const int wave    = blockIdx.x * 8 + w;
    const int base_pt = wave * 4;            // grid sized so every wave is full

    // Stage 1: coalesced b128 staging of 4 anchors x 64 regs = 256 floats.
    {
        const float4* gsrc = (const float4*)(regs + (size_t)base_pt * 64);
        const float4 va = gsrc[lane * 2 + 0];
        const float4 vb = gsrc[lane * 2 + 1];
        *(float4*)&sraw[w][lane * 8 + 0] = va;
        *(float4*)&sraw[w][lane * 8 + 4] = vb;
    }
    __syncthreads();

    // Stage 2: softmax-16 per distribution row.
    if (lane < 16) {
        const float* src = &sraw[w][lane * 16];
        float x[16];
        float mx = -INFINITY;
        #pragma unroll
        for (int k = 0; k < 16; ++k) { x[k] = src[k]; mx = fmaxf(mx, x[k]); }
        float sum = 0.f;
        #pragma unroll
        for (int k = 0; k < 16; ++k) { x[k] = expf(x[k] - mx); sum += x[k]; }
        const float inv = 1.f / sum;
        #pragma unroll
        for (int k = 0; k < 16; ++k) sprob[w][lane][k] = (_Float16)(x[k] * inv);
    }
    __syncthreads();

    // Stage 3: build fragments per ISA 16-bit A layout:
    // lane<16 : K = 0..7 in halves 0..7 (K16..23 zero in halves 8..15)
    // lane>=16: K = 8..15 in halves 0..7 (K24..31 zero)
    const int row   = lane & 15;
    const int kbase = (lane >> 4) * 8;
    v16h A, Bm;
    #pragma unroll
    for (int i = 0; i < 8; ++i)  A[i] = sprob[w][row][kbase + i];
    #pragma unroll
    for (int i = 8; i < 16; ++i) A[i] = (_Float16)0.0f;
    // B fragment: same K order as A: halves 0..7 = proj[kbase+i]=kbase+i; halves 8..15 (K16..31) zero.
    #pragma unroll
    for (int i = 0; i < 16; ++i) Bm[i] = (i < 8) ? (_Float16)(float)(kbase + i) : (_Float16)0.0f;

    v8f C = {};
    C = __builtin_amdgcn_wmma_f32_16x16x32_f16(false, A, false, Bm,
                                               (short)0, C, false, false);

    // C/D layout: lane<16 -> rows 0..7; lane>=16 -> rows 8..15.
    // Every column is identical; lanes 0 and 16 finalize 2 anchors each.
    if ((lane & 15) == 0) {
        const int abase = (lane >> 4) * 2;   // anchors {0,1} or {2,3} of the group
        #pragma unroll
        for (int half = 0; half < 2; ++half) {
            const int pt = base_pt + abase + half;
            const int n  = pt % NN;
            const float d0 = C[half * 4 + 0];
            const float d1 = C[half * 4 + 1];
            const float d2 = C[half * 4 + 2];
            const float d3 = C[half * 4 + 3];
            const float ax = anchors[2 * n], ay = anchors[2 * n + 1];
            const float st = strides[n];
            float* o = boxes + (size_t)pt * 4;
            o[0] = ax - d0 * st;
            o[1] = ay - d1 * st;
            o[2] = ax + d2 * st;
            o[3] = ay + d3 * st;
        }
    }
}

// ---------------------------------------------------------------------------
// Kernel 2: align[b,m,n] = sigmoid(score[b,n,lab]) * iou^6 * in_gts
// ---------------------------------------------------------------------------
__global__ void align_kernel(const float* __restrict__ scores,
                             const float* __restrict__ boxes,
                             const float* __restrict__ anchors,
                             const int*   __restrict__ gt_labels,
                             const float* __restrict__ gt_bboxes,
                             float*       __restrict__ align)
{
    const int bm = blockIdx.x;
    const int b  = bm / MM;
    const float g0 = gt_bboxes[bm * 4 + 0], g1 = gt_bboxes[bm * 4 + 1];
    const float g2 = gt_bboxes[bm * 4 + 2], g3 = gt_bboxes[bm * 4 + 3];
    const int   lab = gt_labels[bm];
    const float ga  = (g2 - g0) * (g3 - g1);

    for (int n = threadIdx.x; n < NN; n += blockDim.x) {
        const float* p = boxes + ((size_t)b * NN + n) * 4;
        const float p0 = p[0], p1 = p[1], p2 = p[2], p3 = p[3];
        const float iw = fminf(p2, g2) - fmaxf(p0, g0);
        const float ih = fminf(p3, g3) - fmaxf(p1, g1);
        const float inter = fmaxf(iw, 0.f) * fmaxf(ih, 0.f);
        const float pa = (p2 - p0) * (p3 - p1);
        const float iou = inter / (pa + ga - inter + EPSF);
        const float ax = anchors[2 * n], ay = anchors[2 * n + 1];
        const float ing = (ax >= g0 && ax <= g2 && ay >= g1 && ay <= g3) ? 1.f : 0.f;
        const float sc = scores[((size_t)b * NN + n) * NCC + lab];
        const float sg = 1.f / (1.f + expf(-sc));
        const float i2 = iou * iou;
        align[(size_t)bm * NN + n] = sg * (i2 * i2 * i2) * ing;
    }
}

// ---------------------------------------------------------------------------
// Kernel 3: iterative top-k per (b,m) row (lowest-index tie-break)
// ---------------------------------------------------------------------------
__global__ void topk_kernel(const float* __restrict__ align,
                            int* __restrict__ topk_idx, int K)
{
    __shared__ float svals[TPB];
    __shared__ int   sidx[TPB];
    __shared__ int   sel[16];
    const int bm = blockIdx.x;
    const float* rowp = align + (size_t)bm * NN;

    for (int k = 0; k < K; ++k) {
        float bv = -INFINITY;
        int   bi = 0x7fffffff;
        for (int n = threadIdx.x; n < NN; n += blockDim.x) {
            bool skip = false;
            for (int j = 0; j < k; ++j) if (sel[j] == n) { skip = true; break; }
            if (skip) continue;
            const float v = rowp[n];
            if (v > bv) { bv = v; bi = n; }       // ascending n: first max kept
        }
        svals[threadIdx.x] = bv; sidx[threadIdx.x] = bi;
        __syncthreads();
        for (int off = TPB / 2; off > 0; off >>= 1) {
            if (threadIdx.x < off) {
                const float ov = svals[threadIdx.x + off];
                const int   oi = sidx[threadIdx.x + off];
                if (ov > svals[threadIdx.x] ||
                    (ov == svals[threadIdx.x] && oi < sidx[threadIdx.x])) {
                    svals[threadIdx.x] = ov; sidx[threadIdx.x] = oi;
                }
            }
            __syncthreads();
        }
        if (threadIdx.x == 0) { sel[k] = sidx[0]; topk_idx[bm * K + k] = sidx[0]; }
        __syncthreads();
    }
}

// ---------------------------------------------------------------------------
// Kernel 4: per-anchor targets. mask_single leaves <=1 nonzero per anchor, so
// mpos = (n in topk[argmax_m align]) * mask_gt[argmax_m].
// ---------------------------------------------------------------------------
__global__ void targets_kernel(const float* __restrict__ align,
                               const int*   __restrict__ topk_idx,
                               const int*   __restrict__ gt_labels,
                               const float* __restrict__ gt_bboxes,
                               const float* __restrict__ mask_gt,
                               int K,
                               int*   __restrict__ tlab,
                               float* __restrict__ tbox,
                               float* __restrict__ mpos)
{
    const int idx = blockIdx.x * blockDim.x + threadIdx.x;
    if (idx >= BB * NN) return;
    const int b = idx / NN, n = idx % NN;

    float bv = -INFINITY; int mi = 0;
    for (int m = 0; m < MM; ++m) {
        const float v = align[((size_t)b * MM + m) * NN + n];
        if (v > bv) { bv = v; mi = m; }           // first max (argmax semantics)
    }
    bool contains = false;
    const int* selp = topk_idx + ((size_t)b * MM + mi) * K;
    for (int j = 0; j < K; ++j) if (selp[j] == n) { contains = true; break; }

    const float mg = mask_gt[b * MM + mi];
    const float mp = contains ? mg : 0.f;
    const int tgi = (mp > 0.f) ? mi : 0;
    tlab[idx] = (mp > 0.f) ? gt_labels[b * MM + tgi] : 0;
    const float* gb = gt_bboxes + ((size_t)b * MM + tgi) * 4;
    #pragma unroll
    for (int c = 0; c < 4; ++c) tbox[(size_t)idx * 4 + c] = gb[c] * mp;
    mpos[idx] = mp;
}

// ---------------------------------------------------------------------------
// Kernel 5: full BCE positive term: sum max(x,0)+log1p(exp(-|x|))
// ---------------------------------------------------------------------------
__global__ void bce_sum_kernel(const float* __restrict__ scores,
                               float* __restrict__ acc)
{
    __shared__ float red[TPB];
    const int total  = BB * NN * NCC;
    const int stride = gridDim.x * blockDim.x;
    float s = 0.f;
    for (int i = blockIdx.x * blockDim.x + threadIdx.x; i < total; i += stride) {
        __builtin_prefetch(&scores[i + stride], 0, 1);   // global_prefetch_b8
        const float x = scores[i];
        s += fmaxf(x, 0.f) + log1pf(expf(-fabsf(x)));
    }
    red[threadIdx.x] = s;
    __syncthreads();
    for (int off = TPB / 2; off > 0; off >>= 1) {
        if (threadIdx.x < off) red[threadIdx.x] += red[threadIdx.x + off];
        __syncthreads();
    }
    if (threadIdx.x == 0) atomicAdd(acc, red[0]);
}

// ---------------------------------------------------------------------------
// Kernel 6: per-anchor loss terms (cls subtraction, CIoU box loss, DFL)
// ---------------------------------------------------------------------------
__global__ void anchor_loss_kernel(const float* __restrict__ scores,
                                   const float* __restrict__ regs,
                                   const float* __restrict__ boxes,
                                   const float* __restrict__ anchors,
                                   const float* __restrict__ strides,
                                   const int*   __restrict__ tlab,
                                   const float* __restrict__ tbox,
                                   const float* __restrict__ mpos,
                                   float* __restrict__ acc)
{
    __shared__ float r0[TPB], r1[TPB], r2[TPB], r3[TPB], r4[TPB];
    const int idx = blockIdx.x * blockDim.x + threadIdx.x;
    float cls_sub = 0.f, box_s = 0.f, dfl_s = 0.f, nfg = 0.f, mps = 0.f;

    if (idx < BB * NN) {
        const int n = idx % NN;
        const float mp = mpos[idx];
        const float fg = (mp > 0.f) ? 1.f : 0.f;
        mps = mp; nfg = fg;
        if (fg > 0.f) {
            const float* p = boxes + (size_t)idx * 4;
            const float* t = tbox  + (size_t)idx * 4;
            const float p0 = p[0], p1 = p[1], p2 = p[2], p3 = p[3];
            const float t0 = t[0], t1 = t[1], t2 = t[2], t3 = t[3];
            const float w1 = p2 - p0, h1 = p3 - p1;
            const float w2 = t2 - t0, h2 = t3 - t1;
            const float iw = fminf(p2, t2) - fmaxf(p0, t0);
            const float ih = fminf(p3, t3) - fmaxf(p1, t1);
            const float inter = fmaxf(iw, 0.f) * fmaxf(ih, 0.f);
            const float uni = w1 * h1 + w2 * h2 - inter + EPSF;
            const float iou = inter / uni;

            // cls: subtract x * ts where ts = iou_p * fg at class tlab
            const float x = scores[(size_t)idx * NCC + tlab[idx]];
            cls_sub = x * iou;

            // CIoU (replicates reference, incl. ch = max(y2s) - b1y1 quirk)
            const float cw = fmaxf(p2, t2) - fminf(p0, t0);
            const float ch = fmaxf(p3, t3) - p1;
            const float c2 = cw * cw + ch * ch + EPSF;
            const float dx = p0 + p2 - t0 - t2;
            const float dy = p1 + p3 - t1 - t3;
            const float rho2 = (dx * dx + dy * dy) * 0.25f;
            const float dv = atanf(w2 / (h2 + EPSF)) - atanf(w1 / (h1 + EPSF));
            const float v  = (4.f / 9.869604401089358f) * dv * dv;
            const float alpha = v / (v - iou + 1.f + EPSF);
            const float ciou = iou - (rho2 / c2 + v * alpha);
            box_s = 1.f - ciou;

            // DFL
            const float ax = anchors[2 * n], ay = anchors[2 * n + 1];
            const float st = strides[n];
            float tt[4];
            tt[0] = (ax - t0) / st; tt[1] = (ay - t1) / st;
            tt[2] = (t2 - ax) / st; tt[3] = (t3 - ay) / st;
            #pragma unroll
            for (int s4 = 0; s4 < 4; ++s4) {
                const float tv = fminf(fmaxf(tt[s4], 0.f), (float)RMM - 1.01f);
                const int   tl = (int)tv;
                const int   tr = tl + 1;
                const float wl = (float)tr - tv;
                const float wr = 1.f - wl;
                const float* rr = regs + (size_t)idx * 64 + s4 * 16;
                float mx = -INFINITY;
                for (int k = 0; k < 16; ++k) mx = fmaxf(mx, rr[k]);
                float se = 0.f;
                for (int k = 0; k < 16; ++k) se += expf(rr[k] - mx);
                const float lse = logf(se);
                const float lpl = rr[tl] - mx - lse;
                const float lpr = rr[tr] - mx - lse;
                dfl_s += -(lpl * wl + lpr * wr);
            }
        }
    }

    r0[threadIdx.x] = cls_sub; r1[threadIdx.x] = box_s; r2[threadIdx.x] = dfl_s;
    r3[threadIdx.x] = nfg;     r4[threadIdx.x] = mps;
    __syncthreads();
    for (int off = TPB / 2; off > 0; off >>= 1) {
        if (threadIdx.x < off) {
            r0[threadIdx.x] += r0[threadIdx.x + off];
            r1[threadIdx.x] += r1[threadIdx.x + off];
            r2[threadIdx.x] += r2[threadIdx.x + off];
            r3[threadIdx.x] += r3[threadIdx.x + off];
            r4[threadIdx.x] += r4[threadIdx.x + off];
        }
        __syncthreads();
    }
    if (threadIdx.x == 0) {
        atomicAdd(acc + 1, r0[0]);
        atomicAdd(acc + 2, r1[0]);
        atomicAdd(acc + 3, r2[0]);
        atomicAdd(acc + 4, r3[0]);
        atomicAdd(acc + 5, r4[0]);
    }
}

__global__ void zero_acc_kernel(float* acc)
{
    if (threadIdx.x < 16) acc[threadIdx.x] = 0.f;
}

__global__ void finalize_kernel(const float* __restrict__ acc,
                                float* __restrict__ out)
{
    if (threadIdx.x != 0 || blockIdx.x != 0) return;
    float tot[2], cl[2], bx[2], df[2];
    for (int br = 0; br < 2; ++br) {
        const float* a = acc + br * 8;
        const float nfg = fmaxf(a[4], 1.f);
        const float mps = fmaxf(a[5], 1.f);
        cl[br] = (a[0] - a[1]) / mps;
        bx[br] = a[2] / nfg;
        df[br] = a[3] / (nfg * 4.f);
        tot[br] = cl[br] + 7.5f * bx[br] + 1.5f * df[br];
    }
    out[0] = tot[0] + tot[1];
    out[1] = cl[0] + cl[1];
    out[2] = bx[0] + bx[1];
    out[3] = df[0] + df[1];
    out[4] = tot[0];
    out[5] = tot[1];
}

extern "C" void kernel_launch(void* const* d_in, const int* in_sizes, int n_in,
                              void* d_out, int out_size, void* d_ws, size_t ws_size,
                              hipStream_t stream)
{
    (void)in_sizes; (void)n_in; (void)out_size; (void)ws_size;
    const float* cls0      = (const float*)d_in[0];
    const float* reg0      = (const float*)d_in[1];
    const float* cls1      = (const float*)d_in[2];
    const float* reg1      = (const float*)d_in[3];
    const float* anchors   = (const float*)d_in[4];
    const float* strides   = (const float*)d_in[5];
    const int*   gt_labels = (const int*)d_in[6];
    const float* gt_bboxes = (const float*)d_in[7];
    const float* mask_gt   = (const float*)d_in[8];

    char* ws = (char*)d_ws;
    size_t off = 0;
    float* boxes = (float*)(ws + off); off += (size_t)BB * NN * 4 * sizeof(float);
    float* align = (float*)(ws + off); off += (size_t)BB * MM * NN * sizeof(float);
    int*   topki = (int*)(ws + off);   off += (size_t)BB * MM * 10 * sizeof(int);
    int*   tlab  = (int*)(ws + off);   off += (size_t)BB * NN * sizeof(int);
    float* tbox  = (float*)(ws + off); off += (size_t)BB * NN * 4 * sizeof(float);
    float* mpos  = (float*)(ws + off); off += (size_t)BB * NN * sizeof(float);
    float* acc   = (float*)(ws + off); off += 16 * sizeof(float);

    zero_acc_kernel<<<1, 32, 0, stream>>>(acc);

    const float* clsA[2] = {cls0, cls1};
    const float* regA[2] = {reg0, reg1};
    const int    Ks[2]   = {10, 1};

    const int decode_blocks  = (BB * NN / 4) / 8;          // 4200, all waves full
    const int anchor_blocks  = (BB * NN + TPB - 1) / TPB;  // 525

    for (int br = 0; br < 2; ++br) {
        decode_wmma_kernel<<<decode_blocks, 256, 0, stream>>>(regA[br], anchors,
                                                              strides, boxes);
        align_kernel<<<BB * MM, TPB, 0, stream>>>(clsA[br], boxes, anchors,
                                                  gt_labels, gt_bboxes, align);
        topk_kernel<<<BB * MM, TPB, 0, stream>>>(align, topki, Ks[br]);
        targets_kernel<<<anchor_blocks, TPB, 0, stream>>>(align, topki, gt_labels,
                                                          gt_bboxes, mask_gt, Ks[br],
                                                          tlab, tbox, mpos);
        bce_sum_kernel<<<2048, TPB, 0, stream>>>(clsA[br], acc + br * 8);
        anchor_loss_kernel<<<anchor_blocks, TPB, 0, stream>>>(clsA[br], regA[br],
                                                              boxes, anchors, strides,
                                                              tlab, tbox, mpos,
                                                              acc + br * 8);
    }
    finalize_kernel<<<1, 1, 0, stream>>>(acc, (float*)d_out);
}
